// catcher_47201690583533
// MI455X (gfx1250) — compile-verified, matrix operand order checked by reading, not run on hardware
//
#include <hip/hip_runtime.h>
#include <math.h>

// ---------------------------------------------------------------------------
// Problem constants
// ---------------------------------------------------------------------------
#define BB   32
#define NN_  288
#define CC   2048
#define RQ   (BB * NN_)            // 9216
#define SCALE_F 0.022097086912079612f  // 2048^-0.5
#define ALD  40                    // LDS row pitch in bf16 (32 data + 8 pad = 80B)

typedef __attribute__((ext_vector_type(16))) __bf16 v16bf;
typedef __attribute__((ext_vector_type(8)))  float  v8f;
typedef unsigned int u32x4 __attribute__((ext_vector_type(4)));
typedef int          i32x8 __attribute__((ext_vector_type(8)));
typedef int          i32x4 __attribute__((ext_vector_type(4)));

#if defined(__has_builtin)
#if __has_builtin(__builtin_amdgcn_tensor_load_to_lds)
#define HAVE_TDM 1
#endif
#endif

__device__ __forceinline__ v8f wmma_bf16(v16bf a, v16bf b, v8f c) {
  return __builtin_amdgcn_wmma_f32_16x16x32_bf16(false, a, false, b,
                                                 (short)0, c, false, false);
}

// ---------------------------------------------------------------------------
// Fragment helpers (ISA 7.12.2 layouts)
// A/B frag: lane r = row/col (0..15), h = lane>>4; elems 0..7  = K h*8..h*8+7,
//                                            elems 8..15 = K 16+h*8..23+h*8
// ---------------------------------------------------------------------------
__device__ __forceinline__ v16bf frag_rows_f32(const float* rowk, int h) {
  v16bf f;
  const float* p0 = rowk + h * 8;
  const float* p1 = rowk + 16 + h * 8;
#pragma unroll
  for (int i = 0; i < 8; ++i) f[i] = (__bf16)p0[i];
#pragma unroll
  for (int i = 0; i < 8; ++i) f[8 + i] = (__bf16)p1[i];
  return f;
}

__device__ __forceinline__ v16bf frag_rows_bf16(const __bf16* rowk, int h) {
  v16bf f;
  const __bf16* p0 = rowk + h * 8;
  const __bf16* p1 = rowk + 16 + h * 8;
#pragma unroll
  for (int i = 0; i < 8; ++i) f[i] = p0[i];
#pragma unroll
  for (int i = 0; i < 8; ++i) f[8 + i] = p1[i];
  return f;
}

__device__ __forceinline__ v16bf frag_col_f32(const float* src, int ld,
                                              int k0, int n, int h) {
  v16bf f;
#pragma unroll
  for (int j = 0; j < 4; ++j) {
    int k = k0 + h * 8 + 2 * j;
    f[2 * j]     = (__bf16)src[(size_t)k * ld + n];
    f[2 * j + 1] = (__bf16)src[(size_t)(k + 1) * ld + n];
  }
#pragma unroll
  for (int j = 0; j < 4; ++j) {
    int k = k0 + 16 + h * 8 + 2 * j;
    f[8 + 2 * j] = (__bf16)src[(size_t)k * ld + n];
    f[9 + 2 * j] = (__bf16)src[(size_t)(k + 1) * ld + n];
  }
  return f;
}

__device__ __forceinline__ v16bf frag_col_bf16(const __bf16* src, int ld,
                                               int k0, int n, int h) {
  v16bf f;
#pragma unroll
  for (int j = 0; j < 4; ++j) {
    int k = k0 + h * 8 + 2 * j;
    f[2 * j]     = src[(size_t)k * ld + n];
    f[2 * j + 1] = src[(size_t)(k + 1) * ld + n];
  }
#pragma unroll
  for (int j = 0; j < 4; ++j) {
    int k = k0 + 16 + h * 8 + 2 * j;
    f[8 + 2 * j] = src[(size_t)k * ld + n];
    f[9 + 2 * j] = src[(size_t)(k + 1) * ld + n];
  }
  return f;
}

// C/D: lane r = N col, h = lane>>4; VGPR i -> M row = m0 + h*8 + i
__device__ __forceinline__ void store_c_bf16(__bf16* D, int ld, int m0, int n0,
                                             const v8f& c, int r, int h) {
  int n = n0 + r;
#pragma unroll
  for (int i = 0; i < 8; ++i)
    D[(size_t)(m0 + h * 8 + i) * ld + n] = (__bf16)c[i];
}

__device__ __forceinline__ void store_c_f32_scaled(float* D, int ld, int m0, int n0,
                                                   const v8f& c, int r, int h,
                                                   float scale) {
  int n = n0 + r;
#pragma unroll
  for (int i = 0; i < 8; ++i)
    D[(size_t)(m0 + h * 8 + i) * ld + n] = c[i] * scale;
}

// ===========================================================================
// LDS-staged big GEMM (NT): D(bf16)[M x 2048] = A @ W^T, K=2048.
// WG = 256 thr (8 waves), WG tile 128x128, wave = 64x32 (4x2 frags).
// MODE 0: A = f32 rows (x)             -> k1: rel_q = x Wq^T
// MODE 1: A = bf16 rows + f32 pos rows -> k5: relg_lin = (gc2+pos) Wg^T
// ===========================================================================
template <int MODE>
__global__ void k_big_nt(const void* __restrict__ Avoid,
                         const float* __restrict__ Pos,
                         const float* __restrict__ W,
                         __bf16* __restrict__ D) {
  __shared__ __bf16 As[128 * ALD];
  __shared__ __bf16 Bs[128 * ALD];
  const int tid = threadIdx.x;
  const int wave = tid >> 5, lane = tid & 31;
  const int r = lane & 15, h = lane >> 4;
  const int wm = wave >> 2, wn = wave & 3;   // 2x4 wave grid
  const int m0 = (blockIdx.x / (CC / 128)) * 128;
  const int n0 = (blockIdx.x % (CC / 128)) * 128;

  const int srow = tid >> 1;          // 0..127
  const int shalf = tid & 1;          // 16-element half of the 32-wide k slab

  v8f acc[4][2];
#pragma unroll
  for (int i = 0; i < 4; ++i)
#pragma unroll
    for (int j = 0; j < 2; ++j) acc[i][j] = (v8f){};

  for (int k0 = 0; k0 < CC; k0 += 32) {
    // ---- stage A tile 128x32 ----
    {
      __bf16* dst = &As[srow * ALD + shalf * 16];
      if (MODE == 0) {
        const float* src = (const float*)Avoid + (size_t)(m0 + srow) * CC + k0 + shalf * 16;
#pragma unroll
        for (int i = 0; i < 16; ++i) dst[i] = (__bf16)src[i];
      } else {
        const __bf16* src = (const __bf16*)Avoid + (size_t)(m0 + srow) * CC + k0 + shalf * 16;
        const float* ps = Pos + (size_t)((m0 + srow) % NN_) * CC + k0 + shalf * 16;
#pragma unroll
        for (int i = 0; i < 16; ++i) dst[i] = (__bf16)((float)src[i] + ps[i]);
      }
    }
    // ---- stage B tile 128x32 (rows of W) ----
    {
      const float* src = W + (size_t)(n0 + srow) * CC + k0 + shalf * 16;
      __bf16* dst = &Bs[srow * ALD + shalf * 16];
#pragma unroll
      for (int i = 0; i < 16; ++i) dst[i] = (__bf16)src[i];
    }
    __syncthreads();

    v16bf bf0 = frag_rows_bf16(&Bs[(wn * 32 + r) * ALD], h);
    v16bf bf1 = frag_rows_bf16(&Bs[(wn * 32 + 16 + r) * ALD], h);
#pragma unroll
    for (int i = 0; i < 4; ++i) {
      v16bf af = frag_rows_bf16(&As[(wm * 64 + i * 16 + r) * ALD], h);
      acc[i][0] = wmma_bf16(af, bf0, acc[i][0]);
      acc[i][1] = wmma_bf16(af, bf1, acc[i][1]);
    }
    __syncthreads();
  }
#pragma unroll
  for (int i = 0; i < 4; ++i)
#pragma unroll
    for (int j = 0; j < 2; ++j)
      store_c_bf16(D, CC, m0 + wm * 64 + i * 16, n0 + wn * 32 + j * 16,
                   acc[i][j], r, h);
}

// ===========================================================================
// Row inverse-norms of g
// ===========================================================================
__global__ void k2_rownorm(const float* __restrict__ G, float* __restrict__ invn) {
  __shared__ float red[256];
  const float* p = G + (size_t)blockIdx.x * CC;
  float s = 0.f;
  for (int i = threadIdx.x; i < CC; i += 256) { float v = p[i]; s += v * v; }
  red[threadIdx.x] = s;
  __syncthreads();
  for (int st = 128; st > 0; st >>= 1) {
    if (threadIdx.x < st) red[threadIdx.x] += red[threadIdx.x + st];
    __syncthreads();
  }
  if (threadIdx.x == 0)
    invn[blockIdx.x] = 1.0f / fmaxf(sqrtf(red[0]), 1e-12f);
}

// ===========================================================================
// S1 = (g g^T) * invn[p]*invn[q] per batch (NT, direct loads, 11 GFLOP)
// ===========================================================================
__global__ void k3_gg(const float* __restrict__ Gall, const float* __restrict__ invn,
                      float* __restrict__ Sall) {
  const int m = blockIdx.y;
  const float* G  = Gall + (size_t)m * NN_ * CC;
  const float* iv = invn + (size_t)m * NN_;
  float* S        = Sall + (size_t)m * NN_ * NN_;
  const int wave = threadIdx.x >> 5, lane = threadIdx.x & 31;
  const int r = lane & 15, h = lane >> 4;
  int tile = blockIdx.x * 8 + wave;
  if (tile >= 81) return;
  int m0 = (tile / 9) * 32, n0 = (tile % 9) * 32;
  v8f c00 = {}, c01 = {}, c10 = {}, c11 = {};
  for (int k0 = 0; k0 < CC; k0 += 32) {
    v16bf a0 = frag_rows_f32(G + (size_t)(m0 + r) * CC + k0, h);
    v16bf a1 = frag_rows_f32(G + (size_t)(m0 + 16 + r) * CC + k0, h);
    v16bf b0 = frag_rows_f32(G + (size_t)(n0 + r) * CC + k0, h);
    v16bf b1 = frag_rows_f32(G + (size_t)(n0 + 16 + r) * CC + k0, h);
    c00 = wmma_bf16(a0, b0, c00);  c01 = wmma_bf16(a0, b1, c01);
    c10 = wmma_bf16(a1, b0, c10);  c11 = wmma_bf16(a1, b1, c11);
  }
#pragma unroll
  for (int i = 0; i < 8; ++i) {
    int mr0 = m0 + h * 8 + i, mr1 = m0 + 16 + h * 8 + i;
    int nc0 = n0 + r, nc1 = n0 + 16 + r;
    S[(size_t)mr0 * NN_ + nc0] = c00[i] * iv[mr0] * iv[nc0];
    S[(size_t)mr0 * NN_ + nc1] = c01[i] * iv[mr0] * iv[nc1];
    S[(size_t)mr1 * NN_ + nc0] = c10[i] * iv[mr1] * iv[nc0];
    S[(size_t)mr1 * NN_ + nc1] = c11[i] * iv[mr1] * iv[nc1];
  }
}

// ===========================================================================
// Row softmax over 288 columns
// ===========================================================================
__global__ void k_softmax(const float* __restrict__ S, float* __restrict__ outf,
                          __bf16* __restrict__ outb) {
  __shared__ float buf[NN_];
  __shared__ float stat;
  const float* p = S + (size_t)blockIdx.x * NN_;
  int t = threadIdx.x;
  float v = p[t];
  buf[t] = v;
  __syncthreads();
  if (t == 0) {
    float mx = -INFINITY;
    for (int i = 0; i < NN_; ++i) mx = fmaxf(mx, buf[i]);
    stat = mx;
  }
  __syncthreads();
  float e = __expf(v - stat);
  buf[t] = e;
  __syncthreads();
  if (t == 0) {
    float s = 0.f;
    for (int i = 0; i < NN_; ++i) s += buf[i];
    stat = s;
  }
  __syncthreads();
  float rres = e / stat;
  if (outf) outf[(size_t)blockIdx.x * NN_ + t] = rres;
  outb[(size_t)blockIdx.x * NN_ + t] = (__bf16)rres;
}

// ===========================================================================
// gc2(bf16) = rel1(bf16) @ g(f32), per m.  K=288 (NN, gathered B)
// ===========================================================================
__global__ void k4_attn_g(const __bf16* __restrict__ Rel1, const float* __restrict__ Gall,
                          __bf16* __restrict__ D) {
  const int m = blockIdx.y;
  const __bf16* A = Rel1 + (size_t)m * NN_ * NN_;
  const float*  G = Gall + (size_t)m * NN_ * CC;
  __bf16*       Dp = D   + (size_t)m * NN_ * CC;
  const int wave = threadIdx.x >> 5, lane = threadIdx.x & 31;
  const int r = lane & 15, h = lane >> 4;
  int tile = blockIdx.x * 8 + wave;
  if (tile >= (NN_ / 32) * (CC / 32)) return;
  int m0 = (tile / (CC / 32)) * 32, n0 = (tile % (CC / 32)) * 32;
  v8f c00 = {}, c01 = {}, c10 = {}, c11 = {};
  for (int k0 = 0; k0 < NN_; k0 += 32) {
    v16bf a0 = frag_rows_bf16(A + (size_t)(m0 + r) * NN_ + k0, h);
    v16bf a1 = frag_rows_bf16(A + (size_t)(m0 + 16 + r) * NN_ + k0, h);
    v16bf b0 = frag_col_f32(G, CC, k0, n0 + r, h);
    v16bf b1 = frag_col_f32(G, CC, k0, n0 + 16 + r, h);
    c00 = wmma_bf16(a0, b0, c00);  c01 = wmma_bf16(a0, b1, c01);
    c10 = wmma_bf16(a1, b0, c10);  c11 = wmma_bf16(a1, b1, c11);
  }
  store_c_bf16(Dp, CC, m0,      n0,      c00, r, h);
  store_c_bf16(Dp, CC, m0,      n0 + 16, c01, r, h);
  store_c_bf16(Dp, CC, m0 + 16, n0,      c10, r, h);
  store_c_bf16(Dp, CC, m0 + 16, n0 + 16, c11, r, h);
}

// ===========================================================================
// TDM descriptor issue: load a [rows x 32] bf16 tile (row pitch 2048 elems)
// into LDS at byte offset ldsOff with 4-dword padding every 16 dwords
// (-> 40-bf16 LDS row pitch = ALD).  6-arg builtin (clang-23 signature).
// ===========================================================================
__device__ __forceinline__ void tdm_load_tile_bf16(const __bf16* gptr,
                                                   unsigned ldsOff, int rows) {
#if defined(HAVE_TDM)
  unsigned long long ga = (unsigned long long)(uintptr_t)gptr;
  u32x4 g0;
  g0[0] = 1u;                                  // count=1, user descriptor
  g0[1] = ldsOff;                              // lds_addr (bytes)
  g0[2] = (unsigned)(ga & 0xffffffffu);        // global_addr[31:0]
  g0[3] = (unsigned)((ga >> 32) & 0x1ffffffu)  // global_addr[56:32]
        | (2u << 30);                          // type = 2 ("image")
  i32x8 g1;
  g1[0] = (int)((1u << 16)      // data_size = 1 -> 2 bytes
              | (1u << 20)      // pad_enable
              | (3u << 22)      // pad_interval code 3 -> every 16 dwords
              | (3u << 25));    // pad_amount  code 3 -> 4 dwords
  g1[1] = (int)((CC & 0xffffu) << 16);             // tensor_dim0[15:0]
  g1[2] = (int)((CC >> 16) | ((RQ & 0xffffu) << 16)); // dim0[31:16] | dim1[15:0]
  g1[3] = (int)((RQ >> 16) | (32u << 16));         // dim1[31:16] | tile_dim0=32
  g1[4] = (int)(unsigned)rows;                     // tile_dim1 | tile_dim2=0
  g1[5] = (int)CC;                                 // tensor_dim0_stride[31:0]
  g1[6] = 0;                                       // stride0[47:32] | stride1 lo
  g1[7] = 0;
  i32x4 z4 = {0, 0, 0, 0};
  i32x8 z8 = {0, 0, 0, 0, 0, 0, 0, 0};
  __builtin_amdgcn_tensor_load_to_lds(g0, g1, z4, z4, z8, 0);
#else
  (void)gptr; (void)ldsOff; (void)rows;
#endif
}

__device__ __forceinline__ void tdm_wait0() {
#if defined(__has_builtin)
#if __has_builtin(__builtin_amdgcn_s_wait_tensorcnt)
  __builtin_amdgcn_s_wait_tensorcnt(0);
#else
  asm volatile("s_wait_tensorcnt 0x0" ::: "memory");
#endif
#else
  asm volatile("s_wait_tensorcnt 0x0" ::: "memory");
#endif
}

// ===========================================================================
// S2 = SCALE * rel_q @ relg_lin^T per b (NT, K=2048), TDM-staged through LDS.
// WG = 288 thr (9 waves, 3x3), WG tile 96x96, wave tile 32x32.
// ===========================================================================
__global__ void k6_qk(const __bf16* __restrict__ Qall, const __bf16* __restrict__ Kall,
                      float* __restrict__ Sall) {
  __shared__ __bf16 smem[2 * 96 * ALD];        // As at LDS off 0, Bs at off 96*ALD*2
  __bf16* As = smem;
  __bf16* Bs = smem + 96 * ALD;
  const unsigned ldsOffA = 0u;
  const unsigned ldsOffB = 96u * ALD * 2u;     // bytes

  const int b = blockIdx.y;
  const __bf16* Q = Qall + (size_t)b * NN_ * CC;
  const __bf16* K = Kall + (size_t)b * NN_ * CC;
  float* S        = Sall + (size_t)b * NN_ * NN_;

  const int tid = threadIdx.x;                 // 288 threads
  const int wave = tid >> 5, lane = tid & 31;
  const int r = lane & 15, h = lane >> 4;
  const int wm = wave / 3, wn = wave % 3;      // 3x3 wave grid
  const int m0 = (blockIdx.x / 3) * 96;
  const int n0 = (blockIdx.x % 3) * 96;

  v8f acc[2][2];
#pragma unroll
  for (int i = 0; i < 2; ++i)
#pragma unroll
    for (int j = 0; j < 2; ++j) acc[i][j] = (v8f){};

  for (int k0 = 0; k0 < CC; k0 += 32) {
#if defined(HAVE_TDM)
    if (wave == 0) {
      tdm_load_tile_bf16(Q + (size_t)m0 * CC + k0, ldsOffA, 96);
      tdm_load_tile_bf16(K + (size_t)n0 * CC + k0, ldsOffB, 96);
      tdm_wait0();
    }
#else
    for (int i = tid; i < 96 * 32; i += 288) {
      int row = i >> 5, col = i & 31;
      As[row * ALD + col] = Q[(size_t)(m0 + row) * CC + k0 + col];
      Bs[row * ALD + col] = K[(size_t)(n0 + row) * CC + k0 + col];
    }
#endif
    __syncthreads();

    v16bf bf0 = frag_rows_bf16(&Bs[(wn * 32 + r) * ALD], h);
    v16bf bf1 = frag_rows_bf16(&Bs[(wn * 32 + 16 + r) * ALD], h);
#pragma unroll
    for (int i = 0; i < 2; ++i) {
      v16bf af = frag_rows_bf16(&As[(wm * 32 + i * 16 + r) * ALD], h);
      acc[i][0] = wmma_bf16(af, bf0, acc[i][0]);
      acc[i][1] = wmma_bf16(af, bf1, acc[i][1]);
    }
    __syncthreads();
  }
#pragma unroll
  for (int i = 0; i < 2; ++i)
#pragma unroll
    for (int j = 0; j < 2; ++j)
      store_c_f32_scaled(S, NN_, m0 + wm * 32 + i * 16, n0 + wn * 32 + j * 16,
                         acc[i][j], r, h, SCALE_F);
}

// ===========================================================================
// out(f32) = rel2(bf16) @ gc2(bf16) per b. K=288 (NN, gathered B)
// ===========================================================================
__global__ void k7_out(const __bf16* __restrict__ Rel2, const __bf16* __restrict__ GC2,
                       float* __restrict__ Out) {
  const int b = blockIdx.y;
  const __bf16* A = Rel2 + (size_t)b * NN_ * NN_;
  const __bf16* G = GC2  + (size_t)b * NN_ * CC;
  float*        O = Out  + (size_t)b * NN_ * CC;
  const int wave = threadIdx.x >> 5, lane = threadIdx.x & 31;
  const int r = lane & 15, h = lane >> 4;
  int tile = blockIdx.x * 8 + wave;
  if (tile >= (NN_ / 32) * (CC / 32)) return;
  int m0 = (tile / (CC / 32)) * 32, n0 = (tile % (CC / 32)) * 32;
  v8f c00 = {}, c01 = {}, c10 = {}, c11 = {};
  for (int k0 = 0; k0 < NN_; k0 += 32) {
    v16bf a0 = frag_rows_bf16(A + (size_t)(m0 + r) * NN_ + k0, h);
    v16bf a1 = frag_rows_bf16(A + (size_t)(m0 + 16 + r) * NN_ + k0, h);
    v16bf b0 = frag_col_bf16(G, CC, k0, n0 + r, h);
    v16bf b1 = frag_col_bf16(G, CC, k0, n0 + 16 + r, h);
    c00 = wmma_bf16(a0, b0, c00);  c01 = wmma_bf16(a0, b1, c01);
    c10 = wmma_bf16(a1, b0, c10);  c11 = wmma_bf16(a1, b1, c11);
  }
  store_c_f32_scaled(O, CC, m0,      n0,      c00, r, h, 1.0f);
  store_c_f32_scaled(O, CC, m0,      n0 + 16, c01, r, h, 1.0f);
  store_c_f32_scaled(O, CC, m0 + 16, n0,      c10, r, h, 1.0f);
  store_c_f32_scaled(O, CC, m0 + 16, n0 + 16, c11, r, h, 1.0f);
}

// ===========================================================================
// Launcher
// ===========================================================================
extern "C" void kernel_launch(void* const* d_in, const int* in_sizes, int n_in,
                              void* d_out, int out_size, void* d_ws, size_t ws_size,
                              hipStream_t stream) {
  const float* x   = (const float*)d_in[0];
  const float* g   = (const float*)d_in[1];
  const float* pos = (const float*)d_in[2];
  const float* Wq  = (const float*)d_in[3];
  const float* Wg  = (const float*)d_in[4];

  float* out_f32  = (float*)d_out;
  float* rel2_f32 = out_f32 + (size_t)BB * NN_ * CC;

  char* w = (char*)d_ws;
  __bf16* relq_b    = (__bf16*)w;  w += (size_t)RQ * CC * 2;
  __bf16* gc2_b     = (__bf16*)w;  w += (size_t)RQ * CC * 2;
  __bf16* relglin_b = (__bf16*)w;  w += (size_t)RQ * CC * 2;
  __bf16* rel1_b    = (__bf16*)w;  w += (size_t)BB * NN_ * NN_ * 2;
  __bf16* rel2_b    = (__bf16*)w;  w += (size_t)BB * NN_ * NN_ * 2;
  float*  Sbuf      = (float*)w;   w += (size_t)BB * NN_ * NN_ * 4;
  float*  invn      = (float*)w;   w += (size_t)RQ * 4;

  const int bigBlocks   = (RQ / 128) * (CC / 128);   // 72*16 = 1152
  const int smallBlocks = (81 + 7) / 8;              // 11
  const int nnBlocks    = (NN_ / 32) * (CC / 32) / 8;// 72

  // 1) rel_q = x @ Wq^T (LDS-staged)
  k_big_nt<0><<<bigBlocks, 256, 0, stream>>>(x, nullptr, Wq, relq_b);
  // 2) row inverse-norms of g
  k2_rownorm<<<RQ, 256, 0, stream>>>(g, invn);
  // 3) S1 = normalized g g^T
  k3_gg<<<dim3(smallBlocks, BB), 256, 0, stream>>>(g, invn, Sbuf);
  // 4) rel1 = softmax(S1)
  k_softmax<<<RQ, NN_, 0, stream>>>(Sbuf, nullptr, rel1_b);
  // 5) gc2 = rel1 @ g
  k4_attn_g<<<dim3(nnBlocks, BB), 256, 0, stream>>>(rel1_b, g, gc2_b);
  // 6) relg_lin = (gc2 + pos) @ Wg^T (LDS-staged, pos folded into staging)
  k_big_nt<1><<<bigBlocks, 256, 0, stream>>>(gc2_b, pos, Wg, relglin_b);
  // 7) S2 = SCALE * rel_q @ relg_lin^T (TDM-staged)
  k6_qk<<<dim3(9, BB), NN_, 0, stream>>>(relq_b, relglin_b, Sbuf);
  // 8) rel2 = softmax(S2) -> f32 output + bf16 operand
  k_softmax<<<RQ, NN_, 0, stream>>>(Sbuf, rel2_f32, rel2_b);
  // 9) out = rel2 @ gc2
  k7_out<<<dim3(nnBlocks, BB), 256, 0, stream>>>(rel2_b, gc2_b, out_f32);
}
